// DTWLosswithoutlabels_57131654971998
// MI455X (gfx1250) — matrix operand order repaired and checked by reading, not compiled, not run on hardware
//
#include <hip/hip_runtime.h>
#include <hip/hip_bf16.h>

typedef __attribute__((ext_vector_type(16))) __bf16 v16bf;
typedef __attribute__((ext_vector_type(8)))  float  v8f;

#define BIGV 1.0e10f
#define NT   384
#define DDIM 512
#define NB   64
#define ROWS_PER_T (NB * NT)           // 24576 rows per tensor
#define NMAT (4 * NB)                  // 256 distance matrices
#define ELEMS_PER_T ((size_t)ROWS_PER_T * DDIM)   // 12582912

// ---------------------------------------------------------------------------
// Kernel 1: inverse L2 norms, one wave per row (512 floats -> 16/lane)
// inv layout: [t][b*384 + r], t: 0=TGT, 1=OTH, 2=X
// ---------------------------------------------------------------------------
__global__ void k_norms(const float* __restrict__ TGT, const float* __restrict__ OTH,
                        const float* __restrict__ X, float* __restrict__ inv) {
  int wave = blockIdx.x * (blockDim.x >> 5) + (threadIdx.x >> 5);
  int lane = threadIdx.x & 31;
  int t = wave / ROWS_PER_T;
  int r = wave - t * ROWS_PER_T;
  const float* src = (t == 0) ? TGT : ((t == 1) ? OTH : X);
  const float* row = src + (size_t)r * DDIM;
  float s = 0.f;
#pragma unroll
  for (int c = 0; c < 4; ++c) {
    const float4* p = (const float4*)(row + c * 128 + lane * 4);
    float4 v = *p;
    s += v.x * v.x + v.y * v.y + v.z * v.z + v.w * v.w;
  }
#pragma unroll
  for (int m = 16; m >= 1; m >>= 1) s += __shfl_xor(s, m, 32);
  if (lane == 0) inv[wave] = rsqrtf(s);
}

// ---------------------------------------------------------------------------
// Kernel 1b: normalize + fp32 -> bf16 hi/lo split, done ONCE per element.
// hi/lo arrays are flat [t][row][k] bf16, row-major — exactly fragment order.
// ---------------------------------------------------------------------------
__global__ void k_split(const float* __restrict__ TGT, const float* __restrict__ OTH,
                        const float* __restrict__ X, const float* __restrict__ inv,
                        __bf16* __restrict__ hi, __bf16* __restrict__ lo) {
  size_t e = ((size_t)blockIdx.x * blockDim.x + threadIdx.x) * 4;
  size_t grow = e >> 9;                       // global row (0..73727)
  int t = (int)(grow / ROWS_PER_T);
  const float* src = (t == 0) ? TGT : ((t == 1) ? OTH : X);
  float4 v = *(const float4*)(src + (e - (size_t)t * ELEMS_PER_T));
  float s = inv[grow];
  float x[4] = {v.x * s, v.y * s, v.z * s, v.w * s};
  union { __bf16 b[4]; uint2 u; } H, L;
#pragma unroll
  for (int i = 0; i < 4; ++i) {
    __bf16 hv = (__bf16)x[i];
    H.b[i] = hv;
    L.b[i] = (__bf16)(x[i] - (float)hv);
  }
  *(uint2*)(hi + e) = H.u;
  *(uint2*)(lo + e) = L.u;
}

// ---------------------------------------------------------------------------
// Kernel 2: distance matrices. Pure load+WMMA K-loop (no conversions).
// Each wave computes a 32x64 tile (2x4 of 16x16 subtiles), 24 WMMAs/K-chunk.
// A hi/lo frags resident; B frags streamed. bf16 hi/lo 3-pass ~ fp32 accuracy.
// mat = p*64 + b; p: 0=(TGT,X) 1=(TGT,TGT) 2=(OTH,X) 3=(OTH,OTH)
// ---------------------------------------------------------------------------
__device__ __forceinline__ v16bf loadA(const __bf16* __restrict__ base, int off, int h) {
  // A 16x32 layout: lane half h holds K = {8h..8h+7} U {16+8h..16+8h+7}
  union { v16bf v; uint4 q[2]; } f;
  f.q[0] = *(const uint4*)(base + off + 8 * h);
  f.q[1] = *(const uint4*)(base + off + 16 + 8 * h);
  return f.v;
}
__device__ __forceinline__ v16bf loadB(const __bf16* __restrict__ base, int off, int h) {
  // B 32x16 layout: lane half h holds K = {8h..8h+7} U {16+8h..16+8h+7}
  union { v16bf v; uint4 q[2]; } f;
  f.q[0] = *(const uint4*)(base + off + 8 * h);
  f.q[1] = *(const uint4*)(base + off + 16 + 8 * h);
  return f.v;
}

__global__ void k_gemm(const __bf16* __restrict__ hiT, const __bf16* __restrict__ loT,
                       float* __restrict__ Dmat) {
  int wid  = blockIdx.x * (blockDim.x >> 5) + (threadIdx.x >> 5);
  int lane = threadIdx.x & 31;
  int mat = wid / 72;                  // 12 (M/32) x 6 (N/64) = 72 tiles/matrix
  int tt  = wid - mat * 72;
  int tm  = tt / 6, tn = tt % 6;
  int p = mat >> 6;
  int b = mat & 63;
  int iA = (p < 2) ? 0 : 1;                              // TGT/OTH
  int iB = (p == 1) ? 0 : ((p == 3) ? 1 : 2);            // TGT/OTH/X

  int nl = lane & 15;
  int h  = lane >> 4;

  const __bf16* baseAh = hiT + (size_t)(iA * ROWS_PER_T + b * NT) * DDIM;
  const __bf16* baseAl = loT + (size_t)(iA * ROWS_PER_T + b * NT) * DDIM;
  const __bf16* baseBh = hiT + (size_t)(iB * ROWS_PER_T + b * NT) * DDIM;
  const __bf16* baseBl = loT + (size_t)(iB * ROWS_PER_T + b * NT) * DDIM;

  int offA[2], offB[4];
#pragma unroll
  for (int mi = 0; mi < 2; ++mi) offA[mi] = (tm * 32 + mi * 16 + nl) * DDIM;
#pragma unroll
  for (int ni = 0; ni < 4; ++ni) offB[ni] = (tn * 64 + ni * 16 + nl) * DDIM;

  v8f acc[2][4] = {};
#pragma unroll 1
  for (int k0 = 0; k0 < DDIM; k0 += 32) {
    v16bf hA[2], lA[2];
#pragma unroll
    for (int mi = 0; mi < 2; ++mi) {
      hA[mi] = loadA(baseAh, offA[mi] + k0, h);
      lA[mi] = loadA(baseAl, offA[mi] + k0, h);
    }
#pragma unroll
    for (int ni = 0; ni < 4; ++ni) {
      v16bf hB = loadB(baseBh, offB[ni] + k0, h);
      v16bf lB = loadB(baseBl, offB[ni] + k0, h);
#pragma unroll
      for (int mi = 0; mi < 2; ++mi) {
        acc[mi][ni] = __builtin_amdgcn_wmma_f32_16x16x32_bf16(
            false, hA[mi], false, hB, (short)0, acc[mi][ni], false, false);
        acc[mi][ni] = __builtin_amdgcn_wmma_f32_16x16x32_bf16(
            false, hA[mi], false, lB, (short)0, acc[mi][ni], false, false);
        acc[mi][ni] = __builtin_amdgcn_wmma_f32_16x16x32_bf16(
            false, lA[mi], false, hB, (short)0, acc[mi][ni], false, false);
      }
    }
  }

  // C/D 16x16 f32 layout: lane (n=l&15, h), VGPR r -> M = r + 8h, N = n
  float* Db = Dmat + (size_t)mat * NT * NT;
#pragma unroll
  for (int mi = 0; mi < 2; ++mi)
#pragma unroll
    for (int ni = 0; ni < 4; ++ni) {
      int rowBase = tm * 32 + mi * 16;
      int colBase = tn * 64 + ni * 16;
#pragma unroll
      for (int r = 0; r < 8; ++r) {
        int row = rowBase + r + 8 * h;
        Db[(size_t)row * NT + colBase + nl] = 1.0f - acc[mi][ni][r];
      }
    }
}

// ---------------------------------------------------------------------------
// Kernel 3: soft-DTW anti-diagonal scan. One block per matrix, thread i = row i.
// Three rotating LDS diagonals; r_k stored in buf[(k+1)%3].
// ---------------------------------------------------------------------------
__global__ void k_dtw(const float* __restrict__ Dmat, float* __restrict__ partial) {
  int mat = blockIdx.x;
  int i = threadIdx.x;
  const float* Drow = Dmat + (size_t)mat * NT * NT + (size_t)i * NT;
  __shared__ float buf[3][NT];
  buf[0][i] = BIGV;                             // r_{-1}
  buf[1][i] = (i == 0) ? Drow[0] : BIGV;        // r_0
  __syncthreads();
  float res = 0.f;
  for (int k = 1; k < 2 * NT - 1; ++k) {
    const float* rp  = buf[k % 3];              // r_{k-1}
    const float* rp2 = buf[(k + 2) % 3];        // r_{k-2}
    float a  = (i > 0) ? rp[i - 1]  : BIGV;
    float bb = rp[i];
    float c  = (i > 0) ? rp2[i - 1] : BIGV;
    float m  = fminf(a, fminf(bb, c));
    float s  = __expf(m - a) + __expf(m - bb) + __expf(m - c);
    float sm = m - __logf(s);                   // gamma = 1
    int j = k - i;
    float dk = (j >= 0 && j < NT) ? Drow[j] : BIGV;
    float rn = dk + sm;
    buf[(k + 1) % 3][i] = rn;
    if (k == 2 * NT - 2 && i == NT - 1) res = rn;
    __syncthreads();
  }
  if (i == NT - 1) partial[mat] = res;
}

// ---------------------------------------------------------------------------
// Kernel 4: deterministic combine (d_xx terms cancel in the reference).
// loss[b] = P0 - 0.5*P1 - P2 + 0.5*P3
// ---------------------------------------------------------------------------
__global__ void k_combine(const float* __restrict__ partial, float* __restrict__ out) {
  int b = threadIdx.x;
  if (b < NB)
    out[b] = partial[b] - 0.5f * partial[64 + b] - partial[128 + b] + 0.5f * partial[192 + b];
}

extern "C" void kernel_launch(void* const* d_in, const int* in_sizes, int n_in,
                              void* d_out, int out_size, void* d_ws, size_t ws_size,
                              hipStream_t stream) {
  (void)in_sizes; (void)n_in; (void)out_size; (void)ws_size;
  const float* TGT = (const float*)d_in[0];
  const float* OTH = (const float*)d_in[1];
  const float* X   = (const float*)d_in[2];
  float* out = (float*)d_out;

  float*  inv     = (float*)d_ws;                        // 73728 floats
  float*  Dmat    = inv + 3 * ROWS_PER_T;                // 256*384*384 floats (151 MB)
  float*  partial = Dmat + (size_t)NMAT * NT * NT;       // 256 floats
  __bf16* hi      = (__bf16*)(partial + 256);            // 3*ELEMS bf16 (75.5 MB)
  __bf16* lo      = hi + 3 * ELEMS_PER_T;                // 3*ELEMS bf16 (75.5 MB)

  // 73728 row-waves, 8 per 256-thread block
  k_norms<<<(3 * ROWS_PER_T) / 8, 256, 0, stream>>>(TGT, OTH, X, inv);
  // 3*24576*512/4 float4-threads / 256
  k_split<<<(int)(3 * ELEMS_PER_T / 4 / 256), 256, 0, stream>>>(TGT, OTH, X, inv, hi, lo);
  // 256 matrices * 72 tiles = 18432 wave-tiles, 8 per block
  k_gemm<<<(NMAT * 72) / 8, 256, 0, stream>>>(hi, lo, Dmat);
  k_dtw<<<NMAT, NT, 0, stream>>>(Dmat, partial);
  k_combine<<<1, 64, 0, stream>>>(partial, out);
}
